// MoF_72713796321645
// MI455X (gfx1250) — compile-verified, matrix operand order checked by reading, not run on hardware
//
#include <hip/hip_runtime.h>
#include <math.h>

typedef __attribute__((ext_vector_type(16))) __bf16 v16bf;
typedef __attribute__((ext_vector_type(8)))  __bf16 v8bf;
typedef __attribute__((ext_vector_type(8)))  float  v8f;

#define D_MODEL 2048
#define NGROUP  8
#define HDG     256
#define KSEL    2
#define HDIM    512            // KSEL * HDG
#define TPB     16             // tokens per block
#define APAD    520            // 512 + 8 bf16 pad (bank-spread for b128 LDS reads)

// ---------------------------------------------------------------------------
// Prep kernel: convert Wm (512x512 f32, row-major [K][N]) into bf16, packed in
// exact WMMA-B register order for v_wmma_f32_16x16x32_bf16:
//   linear o = (((nt*16 + kstep)*32 + lane)*16 + q)
//   kb = (lane>=16) ? 8 : 0
//   K  = kstep*32 + (q<8 ? kb+q : 16+kb+(q-8))
//   N  = nt*16 + (lane&15)
// so each lane's 16 values for one k-step are 32 contiguous bytes.
// ---------------------------------------------------------------------------
__global__ void mof_pack_wm(const float* __restrict__ Wm, __bf16* __restrict__ WmB) {
    int o = blockIdx.x * blockDim.x + threadIdx.x;   // 0 .. 262143
    int q     = o & 15;
    int lane  = (o >> 4) & 31;
    int kstep = (o >> 9) & 15;
    int nt    = o >> 13;
    int kb = (lane & 16) ? 8 : 0;
    int K = kstep * 32 + ((q < 8) ? (kb + q) : (16 + kb + (q - 8)));
    int N = nt * 16 + (lane & 15);
    WmB[o] = (__bf16)Wm[K * HDIM + N];
}

// ---------------------------------------------------------------------------
// Fused MoF kernel: gate (f32) + gather/scale -> LDS bf16 tile + passthrough
// copy + WMMA bf16 GEMM (f32 accum) + routed scatter of results.
// 512 threads = 16 waves; wave w owns token w of the 16-token tile.
// ---------------------------------------------------------------------------
__global__ __launch_bounds__(512, 2)
void mof_kernel(const float* __restrict__ x,
                const float* __restrict__ Wg,
                const float* __restrict__ bm,
                const __bf16* __restrict__ WmB,
                float* __restrict__ out) {
    __shared__ __bf16 hA[TPB * APAD];    // bf16 A tile: 16 tokens x 512 (+pad)
    __shared__ int    sI0[TPB], sI1[TPB];

    const int lane = threadIdx.x & 31;
    const int wv   = threadIdx.x >> 5;              // 0..15 == token in tile
    const long tokbase = (long)blockIdx.x * TPB;

    // ---------------- Phase 1: gate logits (f32, per-wave token) -----------
    const float* xrow = x + (tokbase + wv) * D_MODEL;
    float acc[NGROUP];
#pragma unroll
    for (int g = 0; g < NGROUP; ++g) acc[g] = 0.f;

    for (int i = lane; i < D_MODEL; i += 32) {
        float xv = xrow[i];
        const float4* wg = (const float4*)(Wg + i * NGROUP);
        float4 w0 = wg[0];
        float4 w1 = wg[1];
        acc[0] = fmaf(xv, w0.x, acc[0]);
        acc[1] = fmaf(xv, w0.y, acc[1]);
        acc[2] = fmaf(xv, w0.z, acc[2]);
        acc[3] = fmaf(xv, w0.w, acc[3]);
        acc[4] = fmaf(xv, w1.x, acc[4]);
        acc[5] = fmaf(xv, w1.y, acc[5]);
        acc[6] = fmaf(xv, w1.z, acc[6]);
        acc[7] = fmaf(xv, w1.w, acc[7]);
    }
#pragma unroll
    for (int m = 16; m >= 1; m >>= 1) {
#pragma unroll
        for (int g = 0; g < NGROUP; ++g)
            acc[g] += __shfl_xor(acc[g], m, 32);
    }

    // softmax + top-2 (redundant across lanes; all lanes need G/I locally)
    float mx = acc[0];
#pragma unroll
    for (int g = 1; g < NGROUP; ++g) mx = fmaxf(mx, acc[g]);
    float e[NGROUP], s = 0.f;
#pragma unroll
    for (int g = 0; g < NGROUP; ++g) { e[g] = __expf(acc[g] - mx); s += e[g]; }
    int i0 = 0; float v0 = e[0];
#pragma unroll
    for (int g = 1; g < NGROUP; ++g) if (e[g] > v0) { v0 = e[g]; i0 = g; }
    int i1 = (i0 == 0) ? 1 : 0; float v1 = e[i1];
#pragma unroll
    for (int g = 0; g < NGROUP; ++g)
        if (g != i0 && e[g] > v1) { v1 = e[g]; i1 = g; }
    const float inv = 1.f / s;
    const float G0 = v0 * inv, G1 = v1 * inv;
    if (lane == 0) { sI0[wv] = i0; sI1[wv] = i1; }

    // ---------------- Phase 2: gather+scale -> bf16 A tile in LDS ----------
    __bf16* ha = &hA[wv * APAD];
    for (int j = lane; j < HDIM; j += 32) {
        int  grp = (j < HDG) ? i0 : i1;
        float gv = (j < HDG) ? G0 : G1;
        ha[j] = (__bf16)(gv * xrow[grp * HDG + (j & (HDG - 1))]);
    }

    // ---------------- Phase 3: passthrough copy of unselected groups -------
    float* orow = out + (tokbase + wv) * D_MODEL;
#pragma unroll
    for (int g = 0; g < NGROUP; ++g) {
        if (g == i0 || g == i1) continue;
        const float4* src = (const float4*)(xrow + g * HDG);
        float4*       dst = (float4*)(orow + g * HDG);
        dst[lane]      = src[lane];
        dst[lane + 32] = src[lane + 32];
    }

    __syncthreads();

    // ---------------- Phase 4: WMMA bf16 GEMM [16x512] @ [512x512] ---------
    const int lane15  = lane & 15;
    const int halfsel = lane >> 4;                   // 0 | 1
    const int kb      = halfsel ? 8 : 0;
    const int nt0 = wv * 2, nt1 = wv * 2 + 1;

    const float bv0 = bm[nt0 * 16 + lane15];
    const float bv1 = bm[nt1 * 16 + lane15];
    v8f acc0 = {bv0, bv0, bv0, bv0, bv0, bv0, bv0, bv0};
    v8f acc1 = {bv1, bv1, bv1, bv1, bv1, bv1, bv1, bv1};

    const __bf16* aP  = &hA[lane15 * APAD + kb];
    const __bf16* b0P = WmB + ((size_t)(nt0 * 16) * 32 + lane) * 16;
    const __bf16* b1P = WmB + ((size_t)(nt1 * 16) * 32 + lane) * 16;

#pragma unroll
    for (int k = 0; k < 16; ++k) {
        v8bf alo = *(const v8bf*)(aP + k * 32);       // K = k*32 + kb + 0..7
        v8bf ahi = *(const v8bf*)(aP + k * 32 + 16);  // K = k*32 + 16 + kb + 0..7
        v16bf a = __builtin_shufflevector(alo, ahi,
                    0, 1, 2, 3, 4, 5, 6, 7, 8, 9, 10, 11, 12, 13, 14, 15);
        v16bf b0 = *(const v16bf*)(b0P + k * 512);    // 32B contiguous, pre-packed
        v16bf b1 = *(const v16bf*)(b1P + k * 512);
        acc0 = __builtin_amdgcn_wmma_f32_16x16x32_bf16(
                   false, a, false, b0, (short)0, acc0, false, false);
        acc1 = __builtin_amdgcn_wmma_f32_16x16x32_bf16(
                   false, a, false, b1, (short)0, acc1, false, false);
    }

    // ---------------- Phase 5: routed scatter of D tiles -------------------
    // D layout: lanes 0-15 -> M=r, lanes 16-31 -> M=r+8; N = nt*16 + lane15
#pragma unroll
    for (int r = 0; r < 8; ++r) {
        int t = r + halfsel * 8;
        float* trow = out + (tokbase + t) * D_MODEL;
        {
            int j = nt0 * 16 + lane15;
            int grp = (j < HDG) ? sI0[t] : sI1[t];
            trow[grp * HDG + (j & (HDG - 1))] = acc0[r];
        }
        {
            int j = nt1 * 16 + lane15;
            int grp = (j < HDG) ? sI0[t] : sI1[t];
            trow[grp * HDG + (j & (HDG - 1))] = acc1[r];
        }
    }
}

extern "C" void kernel_launch(void* const* d_in, const int* in_sizes, int n_in,
                              void* d_out, int out_size, void* d_ws, size_t ws_size,
                              hipStream_t stream) {
    const float* x  = (const float*)d_in[0];   // [4,4096,2048] f32
    const float* Wg = (const float*)d_in[1];   // [2048,8] f32
    const float* Wm = (const float*)d_in[2];   // [512,512] f32
    const float* bm = (const float*)d_in[3];   // [512] f32
    float* out = (float*)d_out;                // [4,4096,2048] f32
    __bf16* WmB = (__bf16*)d_ws;               // 512 KB packed bf16 B

    (void)in_sizes; (void)n_in; (void)out_size; (void)ws_size;

    // Pack Wm -> WMMA-B-ordered bf16 (262144 elems)
    mof_pack_wm<<<512, 512, 0, stream>>>(Wm, WmB);

    // Fused MoF: 16384 tokens / 16 per block
    mof_kernel<<<1024, 512, 0, stream>>>(x, Wg, bm, WmB, out);
}
